// SimpleTransformer_29076928594512
// MI455X (gfx1250) — matrix operand order repaired and checked by reading, not compiled, not run on hardware
//
#include <hip/hip_runtime.h>
#include <hip/hip_bf16.h>
#include <math.h>

#define TSEQ 8192
#define DDIM 64
#define NTILE (TSEQ / 16)   // 512 row/col tiles

typedef __attribute__((ext_vector_type(16))) _Float16 v16h;
typedef __attribute__((ext_vector_type(8)))  _Float16 v8h;
typedef __attribute__((ext_vector_type(8)))  float    v8f;

union AB16 { v16h v; v8h c[2]; };

__device__ __forceinline__ v8f wmma_f32_f16(v16h a, v16h b, v8f c) {
  // 8 args: (neg_a, A, neg_b, B, c_mod, C, reuse_a, reuse_b)
  return __builtin_amdgcn_wmma_f32_16x16x32_f16(false, a, false, b, (short)0, c,
                                                false, false);
}

__device__ __forceinline__ void load_kfrag(const _Float16* p, AB16& b0, AB16& b1) {
  b0.c[0] = *(const v8h*)(p);
  b0.c[1] = *(const v8h*)(p + 16);
  b1.c[0] = *(const v8h*)(p + 32);
  b1.c[1] = *(const v8h*)(p + 48);
}

// ---------------------------------------------------------------------------
// Kernel A: embedding gather + Q/K/V projections (y = x @ W^T).
// Q is pre-scaled by log2(e)/sqrt(D) so scores are produced directly in the
// base-2 exponent domain (2^S == e^(QK/sqrt(D))): the hot loops then use the
// raw hardware v_exp_f32 with no log2(e) conversion multiplies.
// ---------------------------------------------------------------------------
__global__ __launch_bounds__(64) void embed_proj_kernel(
    const int* __restrict__ idx, const float* __restrict__ embed_w,
    const float* __restrict__ Wq, const float* __restrict__ Wk,
    const float* __restrict__ Wv,
    _Float16* __restrict__ Qh, _Float16* __restrict__ Kh,
    _Float16* __restrict__ Vth)
{
  const int t = blockIdx.x;
  const int e = threadIdx.x;
  __shared__ float xs[DDIM];
  const int tok = idx[t];
  xs[e] = embed_w[(size_t)tok * DDIM + e];
  __syncthreads();
  const float* wq = Wq + e * DDIM;
  const float* wk = Wk + e * DDIM;
  const float* wv = Wv + e * DDIM;
  float q = 0.f, k = 0.f, v = 0.f;
  #pragma unroll
  for (int d = 0; d < DDIM; ++d) {
    const float xv = xs[d];
    q = fmaf(xv, wq[d], q);
    k = fmaf(xv, wk[d], k);
    v = fmaf(xv, wv[d], v);
  }
  Qh[t * DDIM + e] = (_Float16)(q * (0.125f * 1.4426950408889634f));
  Kh[t * DDIM + e] = (_Float16)k;
  Vth[(size_t)e * TSEQ + t] = (_Float16)v;        // transposed V
}

// ---------------------------------------------------------------------------
// Kernel Z: zero the strictly-upper tile-aligned region with b128 stores.
// ---------------------------------------------------------------------------
__global__ __launch_bounds__(256) void zero_upper_kernel(float* __restrict__ att)
{
  const int r = blockIdx.x;
  const int start = ((r >> 4) + 1) << 4;
  const int n4 = (TSEQ - start) >> 2;
  float4* base = (float4*)(att + (size_t)r * TSEQ + start);
  const float4 z = {0.f, 0.f, 0.f, 0.f};
  for (int k = threadIdx.x; k < n4; k += 256) base[k] = z;
}

// ---------------------------------------------------------------------------
// Kernel B: fused causal attention. One block (8 waves) per 16-row tile.
// ---------------------------------------------------------------------------
__global__ __launch_bounds__(256) void attention_kernel(
    const _Float16* __restrict__ Qh, const _Float16* __restrict__ Kh,
    const _Float16* __restrict__ Vth,
    float* __restrict__ att, float* __restrict__ O_ws,
    float* __restrict__ row_ent, float* __restrict__ row_sum)
{
  const int i    = blockIdx.x;          // query row tile
  const int tid  = threadIdx.x;
  // w is wave-uniform: force it into an SGPR so loop control is scalar.
  const int w    = __builtin_amdgcn_readfirstlane(tid >> 5);
  const int lane = tid & 31;
  const int n    = lane & 15;           // col within 16-tile (also A row)
  const int hi   = lane >> 4;           // high half-group
  const int hi8  = hi * 8;

  __shared__ float    s_l[8][16];
  __shared__ _Float16 s_P[8][16][32];   // per-wave P staging for transpose
  __shared__ float    s_O[8][1024];     // per-wave O for cross-wave reduce
  __shared__ float    s_e[8][16];
  __shared__ float    s_s[8][16];

  // Persistent Q A-fragments (16x32 halves, K=0..31 and K=32..63).
  const _Float16* qrow = Qh + (size_t)(i * 16 + n) * DDIM;
  AB16 qa0, qa1;
  qa0.c[0] = *(const v8h*)(qrow + hi8);
  qa0.c[1] = *(const v8h*)(qrow + 16 + hi8);
  qa1.c[0] = *(const v8h*)(qrow + 32 + hi8);
  qa1.c[1] = *(const v8h*)(qrow + 48 + hi8);

  // Per-lane K-fragment base (row n of tile 0, feature offset hi*8).
  const _Float16* kbase  = Kh + (size_t)n * DDIM + hi * 8;  // tile j at +j*1024
  // Prefetch base: lanes cover 16 rows x 2 cachelines = one full 2KB tile.
  const _Float16* pfbase = Kh + (size_t)n * DDIM + hi * 32;

  // ---------------- pass 1: lane-local exp sums ----------------
  float l[8];
  #pragma unroll
  for (int v = 0; v < 8; ++v) l[v] = 0.f;

  {
    int j = w;
    const _Float16* kp = kbase + (size_t)j * 1024;
    AB16 xa, xb, ya, yb;
    if (j < i) load_kfrag(kp, xa, xb);
    // 2x-unrolled pipeline with alternating buffers (no register copies).
    // Over-reads of tile j+8 (never consumed) stay inside the workspace.
    for (;;) {
      if (j >= i) break;
      load_kfrag(kp + 8192, ya, yb);
      __builtin_prefetch(pfbase + (size_t)(j + 16) * 1024, 0, 3);
      v8f S = {0.f, 0.f, 0.f, 0.f, 0.f, 0.f, 0.f, 0.f};
      S = wmma_f32_f16(qa0.v, xa.v, S);
      S = wmma_f32_f16(qa1.v, xb.v, S);
      #pragma unroll
      for (int v = 0; v < 8; ++v) l[v] += __builtin_amdgcn_exp2f(S[v]);
      j += 8; kp += 8192;
      if (j >= i) break;
      load_kfrag(kp + 8192, xa, xb);
      __builtin_prefetch(pfbase + (size_t)(j + 16) * 1024, 0, 3);
      v8f S2 = {0.f, 0.f, 0.f, 0.f, 0.f, 0.f, 0.f, 0.f};
      S2 = wmma_f32_f16(qa0.v, ya.v, S2);
      S2 = wmma_f32_f16(qa1.v, yb.v, S2);
      #pragma unroll
      for (int v = 0; v < 8; ++v) l[v] += __builtin_amdgcn_exp2f(S2[v]);
      j += 8; kp += 8192;
    }
  }
  // Diagonal tile: owned by wave (i & 7); wave-uniform branch (EXEC all-1s).
  if ((i & 7) == w) {
    AB16 kb0, kb1;
    load_kfrag(kbase + (size_t)i * 1024, kb0, kb1);
    v8f S = {0.f, 0.f, 0.f, 0.f, 0.f, 0.f, 0.f, 0.f};
    S = wmma_f32_f16(qa0.v, kb0.v, S);
    S = wmma_f32_f16(qa1.v, kb1.v, S);
    #pragma unroll
    for (int v = 0; v < 8; ++v) {
      const float e = __builtin_amdgcn_exp2f(S[v]);
      l[v] += (n > v + hi8) ? 0.f : e;           // causal mask
    }
  }

  // Single cross-lane reduction (within each 16-lane half-group).
  #pragma unroll
  for (int v = 0; v < 8; ++v) {
    float x = l[v];
    x += __shfl_xor(x, 1, 32);
    x += __shfl_xor(x, 2, 32);
    x += __shfl_xor(x, 4, 32);
    x += __shfl_xor(x, 8, 32);
    l[v] = x;
  }
  if (n == 0) {
    #pragma unroll
    for (int v = 0; v < 8; ++v) s_l[w][v + hi8] = l[v];
  }
  __syncthreads();

  float Linv[8];
  #pragma unroll
  for (int v = 0; v < 8; ++v) {
    const int r = v + hi8;
    float ll = 0.f;
    #pragma unroll
    for (int ww = 0; ww < 8; ++ww) ll += s_l[ww][r];
    Linv[v] = 1.0f / ll;                 // diagonal element guarantees ll > 0
  }

  // ---------------- pass 2: write P, entropy (log2 domain), P@V ----------
  v8f Oacc[4] = {};
  float ent[8], rsm[8];
  #pragma unroll
  for (int v = 0; v < 8; ++v) { ent[v] = 0.f; rsm[v] = 0.f; }

  const int q = i >> 1;                  // pair index containing the diagonal

  // Interior pairs: both tiles strictly below the diagonal, no masking.
  for (int p = w; p < q; p += 8) {
    const _Float16* kp = kbase + (size_t)(2 * p) * 1024;
    // Full-tile prefetches for the next pair this wave will process.
    __builtin_prefetch(pfbase + (size_t)(2 * (p + 8)) * 1024, 0, 3);
    __builtin_prefetch(pfbase + (size_t)(2 * (p + 8) + 1) * 1024, 0, 3);
    // Load both K tiles up front (single clause, waits far from loads).
    AB16 k0a, k0b, k1a, k1b;
    load_kfrag(kp, k0a, k0b);
    load_kfrag(kp + 1024, k1a, k1b);
    v8f S0 = {0.f, 0.f, 0.f, 0.f, 0.f, 0.f, 0.f, 0.f};
    v8f S1 = {0.f, 0.f, 0.f, 0.f, 0.f, 0.f, 0.f, 0.f};
    S0 = wmma_f32_f16(qa0.v, k0a.v, S0);
    S0 = wmma_f32_f16(qa1.v, k0b.v, S0);
    S1 = wmma_f32_f16(qa0.v, k1a.v, S1);
    S1 = wmma_f32_f16(qa1.v, k1b.v, S1);
    const size_t rbase = (size_t)(i * 16 + hi8) * TSEQ + 2 * p * 16 + n;
    #pragma unroll
    for (int v = 0; v < 8; ++v) {
      const float P0 = __builtin_amdgcn_exp2f(S0[v]) * Linv[v];
      att[rbase + (size_t)v * TSEQ] = P0;
      ent[v] = fmaf(P0, __builtin_amdgcn_logf(P0 + 1e-9f), ent[v]);
      rsm[v] += P0;
      s_P[w][v + hi8][n] = (_Float16)P0;
      const float P1 = __builtin_amdgcn_exp2f(S1[v]) * Linv[v];
      att[rbase + (size_t)v * TSEQ + 16] = P1;
      ent[v] = fmaf(P1, __builtin_amdgcn_logf(P1 + 1e-9f), ent[v]);
      rsm[v] += P1;
      s_P[w][v + hi8][16 + n] = (_Float16)P1;
    }
    const _Float16* pr = &s_P[w][n][0];
    AB16 pa;
    pa.c[0] = *(const v8h*)(pr + hi8);
    pa.c[1] = *(const v8h*)(pr + 16 + hi8);
    const int vrow = p * 32 + hi * 8;
    #pragma unroll
    for (int nt = 0; nt < 4; ++nt) {
      const _Float16* vp = Vth + (size_t)(nt * 16 + n) * TSEQ + vrow;
      AB16 vb;
      vb.c[0] = *(const v8h*)(vp);
      vb.c[1] = *(const v8h*)(vp + 16);
      Oacc[nt] = wmma_f32_f16(pa.v, vb.v, Oacc[nt]);
    }
  }

  // Final pair (diagonal tile + possible straddle): owned by wave (q & 7).
  if ((q & 7) == w) {
    #pragma unroll
    for (int tt = 0; tt < 2; ++tt) {
      const int j = 2 * q + tt;
      if (j <= i) {
        AB16 kb0, kb1;
        load_kfrag(kbase + (size_t)j * 1024, kb0, kb1);
        v8f S = {0.f, 0.f, 0.f, 0.f, 0.f, 0.f, 0.f, 0.f};
        S = wmma_f32_f16(qa0.v, kb0.v, S);
        S = wmma_f32_f16(qa1.v, kb1.v, S);
        const bool diag = (j == i);
        #pragma unroll
        for (int v = 0; v < 8; ++v) {
          const float e = __builtin_amdgcn_exp2f(S[v]) * Linv[v];
          const float P = (diag && n > v + hi8) ? 0.f : e;  // exact zeros
          att[(size_t)(i * 16 + v + hi8) * TSEQ + j * 16 + n] = P;
          ent[v] = fmaf(P, __builtin_amdgcn_logf(P + 1e-9f), ent[v]);
          rsm[v] += P;
          s_P[w][v + hi8][tt * 16 + n] = (_Float16)P;
        }
      } else {
        #pragma unroll
        for (int v = 0; v < 8; ++v)
          s_P[w][v + hi8][tt * 16 + n] = (_Float16)0.f;     // unused K-half
      }
    }
    const _Float16* pr = &s_P[w][n][0];
    AB16 pa;
    pa.c[0] = *(const v8h*)(pr + hi8);
    pa.c[1] = *(const v8h*)(pr + 16 + hi8);
    const int vrow = q * 32 + hi * 8;
    #pragma unroll
    for (int nt = 0; nt < 4; ++nt) {
      const _Float16* vp = Vth + (size_t)(nt * 16 + n) * TSEQ + vrow;
      AB16 vb;
      vb.c[0] = *(const v8h*)(vp);
      vb.c[1] = *(const v8h*)(vp + 16);
      Oacc[nt] = wmma_f32_f16(pa.v, vb.v, Oacc[nt]);
    }
  }

  // Reduce entropy / rowsum partials across the 16-lane group (once).
  #pragma unroll
  for (int v = 0; v < 8; ++v) {
    float e = ent[v], s = rsm[v];
    e += __shfl_xor(e, 1, 32); s += __shfl_xor(s, 1, 32);
    e += __shfl_xor(e, 2, 32); s += __shfl_xor(s, 2, 32);
    e += __shfl_xor(e, 4, 32); s += __shfl_xor(s, 4, 32);
    e += __shfl_xor(e, 8, 32); s += __shfl_xor(s, 8, 32);
    ent[v] = e; rsm[v] = s;
  }
  if (n == 0) {
    #pragma unroll
    for (int v = 0; v < 8; ++v) { s_e[w][v + hi8] = ent[v]; s_s[w][v + hi8] = rsm[v]; }
  }
  #pragma unroll
  for (int v = 0; v < 8; ++v) {
    const int r = v + hi8;
    s_O[w][r * 64 +  0 + n] = Oacc[0][v];
    s_O[w][r * 64 + 16 + n] = Oacc[1][v];
    s_O[w][r * 64 + 32 + n] = Oacc[2][v];
    s_O[w][r * 64 + 48 + n] = Oacc[3][v];
  }
  __syncthreads();

  for (int e = tid; e < 1024; e += 256) {
    float acc = 0.f;
    #pragma unroll
    for (int ww = 0; ww < 8; ++ww) acc += s_O[ww][e];
    O_ws[(size_t)i * 1024 + e] = acc;
  }
  if (tid < 16) {
    float te = 0.f, ts = 0.f;
    #pragma unroll
    for (int ww = 0; ww < 8; ++ww) { te += s_e[ww][tid]; ts += s_s[ww][tid]; }
    // entropy was accumulated in the log2 domain: convert with one ln2 mul.
    row_ent[i * 16 + tid] = -0.6931471805599453f * te;
    row_sum[i * 16 + tid] = ts;
  }
}

// ---------------------------------------------------------------------------
// Kernel C: out = O @ Wout^T
// ---------------------------------------------------------------------------
__global__ __launch_bounds__(64) void out_proj_kernel(
    const float* __restrict__ O, const float* __restrict__ Wout,
    float* __restrict__ outp)
{
  const int t = blockIdx.x;
  const int e = threadIdx.x;
  __shared__ float xs[DDIM];
  xs[e] = O[(size_t)t * DDIM + e];
  __syncthreads();
  const float* wr = Wout + e * DDIM;
  float acc = 0.f;
  #pragma unroll
  for (int d = 0; d < DDIM; ++d) acc = fmaf(xs[d], wr[d], acc);
  outp[(size_t)t * DDIM + e] = acc;
}

// ---------------------------------------------------------------------------
// Kernel D: prefix scans for entropy (cumsum/count) and energy (cumsum).
// ---------------------------------------------------------------------------
__global__ __launch_bounds__(256) void scan_kernel(
    const float* __restrict__ row_ent, const float* __restrict__ row_sum,
    float* __restrict__ entropy, float* __restrict__ energy)
{
  const int C = TSEQ / 256;
  __shared__ float pe[256], ps[256];
  const int tid  = threadIdx.x;
  const int base = tid * C;
  float se = 0.f, ss = 0.f;
  for (int c = 0; c < C; ++c) { se += row_ent[base + c]; ss += row_sum[base + c]; }
  pe[tid] = se; ps[tid] = ss;
  __syncthreads();
  if (tid == 0) {
    float ae = 0.f, as = 0.f;
    for (int k = 0; k < 256; ++k) {
      const float te = pe[k], ts = ps[k];
      pe[k] = ae; ps[k] = as;
      ae += te; as += ts;
    }
  }
  __syncthreads();
  float ce = pe[tid], cs = ps[tid];
  for (int c = 0; c < C; ++c) {
    ce += row_ent[base + c];
    cs += row_sum[base + c];
    entropy[base + c] = ce / (float)(base + c + 1);
    energy[base + c]  = cs;
  }
}

// ---------------------------------------------------------------------------
extern "C" void kernel_launch(void* const* d_in, const int* in_sizes, int n_in,
                              void* d_out, int out_size, void* d_ws, size_t ws_size,
                              hipStream_t stream) {
  const int*   idx     = (const int*)d_in[0];
  const float* embed_w = (const float*)d_in[1];
  const float* Wq      = (const float*)d_in[2];
  const float* Wk      = (const float*)d_in[3];
  const float* Wv      = (const float*)d_in[4];
  const float* Wout    = (const float*)d_in[5];

  float* att     = (float*)d_out;
  float* entropy = att + (size_t)TSEQ * TSEQ;
  float* energy  = entropy + TSEQ;
  float* outp    = energy + TSEQ;

  char* wsb = (char*)d_ws;
  _Float16* Qh  = (_Float16*)(wsb);                       // 1 MB
  _Float16* Kh  = (_Float16*)(wsb + (1u << 20));          // 1 MB
  _Float16* Vth = (_Float16*)(wsb + (2u << 20));          // 1 MB (transposed)
  float*    O   = (float*)   (wsb + (3u << 20));          // 2 MB
  float*    re  = (float*)   (wsb + (5u << 20));          // 32 KB
  float*    rsu = (float*)   (wsb + (5u << 20) + 32768);  // 32 KB

  zero_upper_kernel<<<TSEQ, 256, 0, stream>>>(att);
  embed_proj_kernel<<<TSEQ, 64, 0, stream>>>(idx, embed_w, Wq, Wk, Wv, Qh, Kh, Vth);
  attention_kernel<<<NTILE, 256, 0, stream>>>(Qh, Kh, Vth, att, O, re, rsu);
  out_proj_kernel<<<TSEQ, 64, 0, stream>>>(O, Wout, outp);
  scan_kernel<<<1, 256, 0, stream>>>(re, rsu, entropy, energy);
}
